// BiasedMultiHeadAttention_39960375722281
// MI455X (gfx1250) — compile-verified, matrix operand order checked by reading, not run on hardware
//
#include <hip/hip_runtime.h>

#define BB 4
#define LL 1024
#define EE 1024
#define HH 16
#define DD 64

constexpr float SCALE = 0.125f;   // D^-0.5 = 64^-0.5
constexpr float EPS   = 1e-5f;

typedef __attribute__((ext_vector_type(16))) _Float16 v16h;
typedef __attribute__((ext_vector_type(8)))  _Float16 v8h;
typedef __attribute__((ext_vector_type(8)))  float    v8f;
typedef __attribute__((ext_vector_type(4)))  int      v4i;

#if __has_builtin(__builtin_amdgcn_global_load_async_to_lds_b128)
#define HAS_ASYNC 1
#else
#define HAS_ASYNC 0
#endif

// global_load_async_to_lds_b128 takes (v4i AS1*, v4i AS3*, imm offset, imm cpol)
#define AS1P(p) ((__attribute__((address_space(1))) v4i*)(const_cast<_Float16*>(p)))
#define AS3P(p) ((__attribute__((address_space(3))) v4i*)(p))

template <int N>
__device__ inline void wait_async_le() {
#if HAS_ASYNC
#if __has_builtin(__builtin_amdgcn_s_wait_asynccnt)
  __builtin_amdgcn_s_wait_asynccnt(N);
#else
  asm volatile("s_wait_asynccnt %0" ::"i"(N) : "memory");
#endif
#endif
}

// ---------------------------------------------------------------------------
// 16-bit A/B fragment load for V_WMMA_F32_16X16X32_F16 (wave32).
// Per ISA 7.12.2: lanes 0-15 hold rows with K = k0..k0+7 (VGPR0-3) and
// K = k0+16..k0+23 (VGPR4-7); lanes 16-31 use K base +8.
// ---------------------------------------------------------------------------
__device__ inline v16h load_frag_g(const _Float16* __restrict__ base,
                                   int row0, int ld, int k0, int lane) {
  int r = row0 + (lane & 15);
  int k = k0 + ((lane >> 4) << 3);
  const _Float16* p = base + (size_t)r * ld + k;
  v8h lo = *(const v8h*)p;
  v8h hi = *(const v8h*)(p + 16);
  v16h out;
#pragma unroll
  for (int i = 0; i < 8; ++i) { out[i] = lo[i]; out[i + 8] = hi[i]; }
  return out;
}

// Same fragment pattern from an LDS tile with fixed row stride of 32 halves.
__device__ inline v16h load_frag_s(const _Float16* base, int row0, int lane) {
  int r = row0 + (lane & 15);
  int k = (lane >> 4) << 3;
  const _Float16* p = base + r * 32 + k;
  v8h lo = *(const v8h*)p;        // ds_load_b128
  v8h hi = *(const v8h*)(p + 16); // ds_load_b128
  v16h out;
#pragma unroll
  for (int i = 0; i < 8; ++i) { out[i] = lo[i]; out[i + 8] = hi[i]; }
  return out;
}

__device__ inline v8f wmma_f16(v16h a, v16h b, v8f c) {
  return __builtin_amdgcn_wmma_f32_16x16x32_f16(false, a, false, b,
                                                (short)0, c, false, false);
}

// ---------------------------------------------------------------------------
// f32 -> f16 elementwise (weight conversion)
// ---------------------------------------------------------------------------
__global__ __launch_bounds__(256) void f32_to_f16_kernel(
    const float* __restrict__ src, _Float16* __restrict__ dst, int n) {
  int i = blockIdx.x * 256 + threadIdx.x;
  if (i < n) dst[i] = (_Float16)src[i];
}

// ---------------------------------------------------------------------------
// LayerNorm over E=1024, one row per 256-thread block, f16 output
// ---------------------------------------------------------------------------
__global__ __launch_bounds__(256) void layernorm_kernel(
    const float* __restrict__ x, const float* __restrict__ g,
    const float* __restrict__ beta, _Float16* __restrict__ xn) {
  int row = blockIdx.x;
  const float4* xr = (const float4*)(x + (size_t)row * EE);
  float4 v = xr[threadIdx.x];
  float s  = v.x + v.y + v.z + v.w;
  float s2 = v.x * v.x + v.y * v.y + v.z * v.z + v.w * v.w;
#pragma unroll
  for (int m = 1; m < 32; m <<= 1) {
    s  += __shfl_xor(s,  m, 32);
    s2 += __shfl_xor(s2, m, 32);
  }
  __shared__ float r1[8], r2[8];
  int lane = threadIdx.x & 31, wv = threadIdx.x >> 5;
  if (lane == 0) { r1[wv] = s; r2[wv] = s2; }
  __syncthreads();
  s = 0.0f; s2 = 0.0f;
#pragma unroll
  for (int i = 0; i < 8; ++i) { s += r1[i]; s2 += r2[i]; }
  float mu   = s * (1.0f / EE);
  float var  = s2 * (1.0f / EE) - mu * mu;
  float rstd = rsqrtf(var + EPS);
  float4 gg = ((const float4*)g)[threadIdx.x];
  float4 bb = ((const float4*)beta)[threadIdx.x];
  _Float16* o = xn + (size_t)row * EE + threadIdx.x * 4;
  o[0] = (_Float16)((v.x - mu) * rstd * gg.x + bb.x);
  o[1] = (_Float16)((v.y - mu) * rstd * gg.y + bb.y);
  o[2] = (_Float16)((v.z - mu) * rstd * gg.z + bb.z);
  o[3] = (_Float16)((v.w - mu) * rstd * gg.w + bb.w);
}

// ---------------------------------------------------------------------------
// WMMA GEMM, LDS-staged + double-buffered: out[m,n] = A[m,:].W[n,:] + bvec[n]
// A: MxK f16 row-major, W: NxK f16 row-major (so out = A @ W^T).
// Block = 256 threads = 8 waves; block tile 128(M) x 64(N), BK=32;
// wave tile 32x32 (2x2 16x16x32 WMMA fragments).
// Stage copy uses GLOBAL_LOAD_ASYNC_TO_LDS_B128 (ASYNCcnt) when available.
// mode 0/1 : out f16 [B,H,L,D]      (Q / K)
// mode 2   : out f16 [B,H,D,L]      (V transposed)
// mode 3   : out f32 [M,N] = residual + acc + bvec   (final projection)
// ---------------------------------------------------------------------------
__global__ __launch_bounds__(256) void wmma_gemm_kernel(
    const _Float16* __restrict__ A, const _Float16* __restrict__ W,
    const float* __restrict__ bvec, const float* __restrict__ residual,
    void* __restrict__ out, int mode) {
  __shared__ _Float16 sA[2][128 * 32];   // 8 KB per buffer
  __shared__ _Float16 sB[2][64 * 32];    // 4 KB per buffer

  int lane = threadIdx.x & 31;
  int w    = threadIdx.x >> 5;
  int wm = w >> 1, wn = w & 1;
  int row0 = blockIdx.y * 128;
  int col0 = blockIdx.x * 64;

  int t = threadIdx.x;
  // chunk decode for the stage copy: each 16B chunk = (row, 8-half k offset)
  int ra0 = t >> 2,          ka0 = (t & 3) * 8;          // A chunk 0
  int ra1 = (t + 256) >> 2,  ka1 = ((t + 256) & 3) * 8;  // A chunk 1
  int rb0 = t >> 2,          kb0 = (t & 3) * 8;          // B chunk 0

  auto stage = [&](int s, int buf) {
    int kk = s * 32;
#if HAS_ASYNC
    __builtin_amdgcn_global_load_async_to_lds_b128(
        AS1P(A + (size_t)(row0 + ra0) * EE + kk + ka0),
        AS3P(&sA[buf][ra0 * 32 + ka0]), 0, 0);
    __builtin_amdgcn_global_load_async_to_lds_b128(
        AS1P(A + (size_t)(row0 + ra1) * EE + kk + ka1),
        AS3P(&sA[buf][ra1 * 32 + ka1]), 0, 0);
    __builtin_amdgcn_global_load_async_to_lds_b128(
        AS1P(W + (size_t)(col0 + rb0) * EE + kk + kb0),
        AS3P(&sB[buf][rb0 * 32 + kb0]), 0, 0);
#else
    v8h tA0 = *(const v8h*)(A + (size_t)(row0 + ra0) * EE + kk + ka0);
    v8h tA1 = *(const v8h*)(A + (size_t)(row0 + ra1) * EE + kk + ka1);
    v8h tB0 = *(const v8h*)(W + (size_t)(col0 + rb0) * EE + kk + kb0);
    *(v8h*)&sA[buf][ra0 * 32 + ka0] = tA0;
    *(v8h*)&sA[buf][ra1 * 32 + ka1] = tA1;
    *(v8h*)&sB[buf][rb0 * 32 + kb0] = tB0;
#endif
  };

  v8f acc[2][2] = {};
  constexpr int NS = EE / 32;
  stage(0, 0);
  for (int s = 0; s < NS; ++s) {
    int buf = s & 1;
    if (s + 1 < NS) {
      stage(s + 1, buf ^ 1);
      wait_async_le<3>();   // keep only next stage's 3 copies in flight
    } else {
      wait_async_le<0>();
    }
    __syncthreads();        // everyone's stage-s data is in LDS

    v16h a0 = load_frag_s(&sA[buf][0], wm * 32,      lane);
    v16h a1 = load_frag_s(&sA[buf][0], wm * 32 + 16, lane);
    v16h b0 = load_frag_s(&sB[buf][0], wn * 32,      lane);
    v16h b1 = load_frag_s(&sB[buf][0], wn * 32 + 16, lane);
    acc[0][0] = wmma_f16(a0, b0, acc[0][0]);
    acc[0][1] = wmma_f16(a0, b1, acc[0][1]);
    acc[1][0] = wmma_f16(a1, b0, acc[1][0]);
    acc[1][1] = wmma_f16(a1, b1, acc[1][1]);

    __syncthreads();        // protect buffer before stage s+2 overwrites it
  }

  int nl = lane & 15, hf = lane >> 4;
#pragma unroll
  for (int fm = 0; fm < 2; ++fm)
#pragma unroll
    for (int fn = 0; fn < 2; ++fn)
#pragma unroll
      for (int i = 0; i < 8; ++i) {
        int m = row0 + wm * 32 + fm * 16 + hf * 8 + i;  // C: M = vgpr + 8*half
        int n = col0 + wn * 32 + fn * 16 + nl;          //    N = lane & 15
        float val = acc[fm][fn][i] + bvec[n];
        int b = m >> 10, l = m & (LL - 1);
        int h = n >> 6,  d = n & (DD - 1);
        if (mode <= 1) {
          ((_Float16*)out)[(((size_t)b * HH + h) * LL + l) * DD + d] =
              (_Float16)val;
        } else if (mode == 2) {
          ((_Float16*)out)[(((size_t)b * HH + h) * DD + d) * LL + l] =
              (_Float16)val;
        } else {
          size_t idx = (size_t)m * EE + n;
          __builtin_nontemporal_store(residual[idx] + val, (float*)out + idx);
        }
      }
}

// ---------------------------------------------------------------------------
// Flash attention with bias/gate/mask, online softmax.
// One wave handles 16 query rows of one (b,h). 128 threads = 4 waves/block.
// Q,K: f16 [B,H,L,D];  Vt: f16 [B,H,D,L];  bias: f32 [B,H,L,L] (NT stream).
// Output O16: f16 [B,L,E] (E index = h*D + d), query-mask applied.
// ---------------------------------------------------------------------------
__global__ __launch_bounds__(128) void flash_attn_kernel(
    const _Float16* __restrict__ Q16, const _Float16* __restrict__ K16,
    const _Float16* __restrict__ Vt16, const float* __restrict__ bias,
    const int* __restrict__ mask, const float* __restrict__ gate,
    _Float16* __restrict__ O16) {
  __shared__ _Float16 pbuf[4][16 * 48];   // per-wave P tile, row stride 48
  int lane = threadIdx.x & 31;
  int wv   = threadIdx.x >> 5;
  int tile = blockIdx.x * 4 + wv;         // 0 .. B*H*(L/16)-1 = 4095
  int b   = tile >> 10;
  int rem = tile & 1023;
  int h   = rem >> 6;
  int qb  = (rem & 63) << 4;

  const _Float16* Qh = Q16 + (((size_t)b * HH + h) * LL + qb) * DD;
  const _Float16* Kh = K16 + ((size_t)b * HH + h) * LL * DD;
  const _Float16* Vh = Vt16 + ((size_t)b * HH + h) * DD * LL;
  const float* biasq = bias + (((size_t)b * HH + h) * LL + qb) * LL;
  const int* maskb = mask + (size_t)b * LL;
  float gh = gate[h];

  v16h q0 = load_frag_g(Qh, 0, DD, 0,  lane);
  v16h q1 = load_frag_g(Qh, 0, DD, 32, lane);

  v8f o0 = {}, o1 = {}, o2 = {}, o3 = {};
  float mrow[8], lrow[8];
#pragma unroll
  for (int i = 0; i < 8; ++i) { mrow[i] = -3.0e38f; lrow[i] = 0.0f; }

  int nl = lane & 15, hf = lane >> 4;
  _Float16* prow = &pbuf[wv][0];

  for (int kb = 0; kb < LL; kb += 32) {
    v16h k00 = load_frag_g(Kh, kb,      DD, 0,  lane);
    v16h k01 = load_frag_g(Kh, kb,      DD, 32, lane);
    v16h k10 = load_frag_g(Kh, kb + 16, DD, 0,  lane);
    v16h k11 = load_frag_g(Kh, kb + 16, DD, 32, lane);
    v8f s0 = {}, s1 = {};
    s0 = wmma_f16(q0, k00, s0);  s0 = wmma_f16(q1, k01, s0);
    s1 = wmma_f16(q0, k10, s1);  s1 = wmma_f16(q1, k11, s1);

    float km0 = maskb[kb + nl]      ? 0.0f : -10000.0f;
    float km1 = maskb[kb + 16 + nl] ? 0.0f : -10000.0f;

#pragma unroll
    for (int i = 0; i < 8; ++i) {
      int ro = hf * 8 + i;                       // query row within tile
      const float* bp = biasq + (size_t)ro * LL + kb;
      float bv0 = __builtin_nontemporal_load(bp + nl);       // 256MB stream:
      float bv1 = __builtin_nontemporal_load(bp + 16 + nl);  // keep out of L2
      float t0 = s0[i] * SCALE + gh * bv0 + km0;
      float t1 = s1[i] * SCALE + gh * bv1 + km1;
      float v = fmaxf(t0, t1);
#pragma unroll
      for (int mk = 1; mk < 16; mk <<= 1) v = fmaxf(v, __shfl_xor(v, mk, 32));
      float nm   = fmaxf(mrow[i], v);
      float corr = __expf(mrow[i] - nm);
      float p0 = __expf(t0 - nm);
      float p1 = __expf(t1 - nm);
      float rs = p0 + p1;
#pragma unroll
      for (int mk = 1; mk < 16; mk <<= 1) rs += __shfl_xor(rs, mk, 32);
      lrow[i] = lrow[i] * corr + rs;
      mrow[i] = nm;
      o0[i] = o0[i] * corr;  o1[i] = o1[i] * corr;
      o2[i] = o2[i] * corr;  o3[i] = o3[i] * corr;
      prow[ro * 48 + nl]      = (_Float16)p0;   // stage P (C-frag layout)
      prow[ro * 48 + 16 + nl] = (_Float16)p1;
    }
    __builtin_amdgcn_wave_barrier();   // DS ops from one wave are in-order

    int rr = nl;
    int kc = hf << 3;
    v8h plo = *(const v8h*)&prow[rr * 48 + kc];
    v8h phi = *(const v8h*)&prow[rr * 48 + kc + 16];
    v16h pf;
#pragma unroll
    for (int i = 0; i < 8; ++i) { pf[i] = plo[i]; pf[i + 8] = phi[i]; }
    __builtin_amdgcn_wave_barrier();

    v16h vb0 = load_frag_g(Vh, 0,  LL, kb, lane);
    v16h vb1 = load_frag_g(Vh, 16, LL, kb, lane);
    v16h vb2 = load_frag_g(Vh, 32, LL, kb, lane);
    v16h vb3 = load_frag_g(Vh, 48, LL, kb, lane);
    o0 = wmma_f16(pf, vb0, o0);
    o1 = wmma_f16(pf, vb1, o1);
    o2 = wmma_f16(pf, vb2, o2);
    o3 = wmma_f16(pf, vb3, o3);
  }

#pragma unroll
  for (int i = 0; i < 8; ++i) {
    int ro = hf * 8 + i;
    int q  = qb + ro;
    float qm  = (float)maskb[q];
    float inv = (lrow[i] > 0.0f) ? (qm / lrow[i]) : 0.0f;
    _Float16* op = O16 + ((size_t)b * LL + q) * EE + h * DD + nl;
    op[0]  = (_Float16)(o0[i] * inv);
    op[16] = (_Float16)(o1[i] * inv);
    op[32] = (_Float16)(o2[i] * inv);
    op[48] = (_Float16)(o3[i] * inv);
  }
}

// ---------------------------------------------------------------------------
// Launch
// ---------------------------------------------------------------------------
extern "C" void kernel_launch(void* const* d_in, const int* in_sizes, int n_in,
                              void* d_out, int out_size, void* d_ws,
                              size_t ws_size, hipStream_t stream) {
  const float* x    = (const float*)d_in[0];
  const float* bias = (const float*)d_in[1];
  const int*   mask = (const int*)d_in[2];
  const float* wq = (const float*)d_in[3];
  const float* bq = (const float*)d_in[4];
  const float* wk = (const float*)d_in[5];
  const float* bk = (const float*)d_in[6];
  const float* wv = (const float*)d_in[7];
  const float* bv = (const float*)d_in[8];
  const float* wo = (const float*)d_in[9];
  const float* bo = (const float*)d_in[10];
  const float* gate = (const float*)d_in[11];
  const float* ln_g = (const float*)d_in[12];
  const float* ln_b = (const float*)d_in[13];

  // workspace layout (48 MB total)
  char* p = (char*)d_ws;
  _Float16* xn16 = (_Float16*)p;  p += (size_t)BB * LL * EE * 2;          // 8 MB
  _Float16* wq16 = (_Float16*)p;  p += (size_t)EE * EE * 2;               // 2 MB
  _Float16* wk16 = (_Float16*)p;  p += (size_t)EE * EE * 2;
  _Float16* wv16 = (_Float16*)p;  p += (size_t)EE * EE * 2;
  _Float16* wo16 = (_Float16*)p;  p += (size_t)EE * EE * 2;
  _Float16* Q16  = (_Float16*)p;  p += (size_t)BB * HH * LL * DD * 2;     // 8 MB
  _Float16* K16  = (_Float16*)p;  p += (size_t)BB * HH * LL * DD * 2;
  _Float16* Vt16 = (_Float16*)p;  p += (size_t)BB * HH * LL * DD * 2;
  _Float16* O16  = (_Float16*)p;  p += (size_t)BB * LL * EE * 2;          // 8 MB

  const int nW = EE * EE;
  f32_to_f16_kernel<<<(nW + 255) / 256, 256, 0, stream>>>(wq, wq16, nW);
  f32_to_f16_kernel<<<(nW + 255) / 256, 256, 0, stream>>>(wk, wk16, nW);
  f32_to_f16_kernel<<<(nW + 255) / 256, 256, 0, stream>>>(wv, wv16, nW);
  f32_to_f16_kernel<<<(nW + 255) / 256, 256, 0, stream>>>(wo, wo16, nW);

  layernorm_kernel<<<BB * LL, 256, 0, stream>>>(x, ln_g, ln_b, xn16);

  dim3 ggrid(EE / 64, (BB * LL) / 128);
  wmma_gemm_kernel<<<ggrid, 256, 0, stream>>>(xn16, wq16, bq, nullptr, Q16, 0);
  wmma_gemm_kernel<<<ggrid, 256, 0, stream>>>(xn16, wk16, bk, nullptr, K16, 1);
  wmma_gemm_kernel<<<ggrid, 256, 0, stream>>>(xn16, wv16, bv, nullptr, Vt16, 2);

  // B*H*(L/16) = 4096 query tiles, 4 waves per 128-thread block
  flash_attn_kernel<<<(BB * HH * (LL / 16)) / 4, 128, 0, stream>>>(
      Q16, K16, Vt16, bias, mask, gate, O16);

  wmma_gemm_kernel<<<ggrid, 256, 0, stream>>>(O16, wo16, bo, x, d_out, 3);
}
